// FGGConv2D_7937099563338
// MI455X (gfx1250) — matrix operand order repaired and checked by reading, not compiled, not run on hardware
//
#include <hip/hip_runtime.h>

// FGGConv2D on MI455X (gfx1250): im2col GEMM (N=131072, K=577, M=128).
// V_WMMA_F32_16X16X32_F16 (f16 in, f32 accumulate) at the HBM roofline.
// W is pre-swizzled into WMMA fragment order (coalesced 32B/lane B loads);
// 128-pixel f16 hcat tile in LDS shared by 8 waves; each wave reuses one B
// fragment across 8 row-tiles. k-loop kept rolled (#pragma unroll 1) with a
// manually pipelined B prefetch to stay under 256 VGPRs (round 3's full
// unroll spilled into s_set_vgpr_msb territory and v_nop hazard padding).

typedef _Float16 v16h __attribute__((ext_vector_type(16)));
typedef _Float16 v8h  __attribute__((ext_vector_type(8)));
typedef float    v8f  __attribute__((ext_vector_type(8)));

#define TILE_N   128           // pixels per workgroup (8 row-tiles x 16)
#define NTILES   8             // row-tiles per wave
#define KDIM     577           // real hcat length
#define KPAD     608           // padded: 19 chunks of K=32
#define NCHUNK   19
#define HSTRIDE  616           // LDS hcat row stride (halfs); 308 dwords ->
                               // 16 A-lanes hit 16 distinct 4-bank groups
#define OUTC     128
#define NWAVE    8

// ---- prep: W (128x577 f32) -> Whf in WMMA-fragment order -------------------
// Whf[o_tile][chunk][lane][e] , e=0..15:  n = o_tile*16 + (lane&15),
// k = chunk*32 + 16*(lane>>4) + e   (zero beyond k=576).
__global__ __launch_bounds__(256)
void prep_w_frag(const float* __restrict__ W, _Float16* __restrict__ Whf) {
    const int d = blockIdx.x * 256 + threadIdx.x;      // dest half index
    if (d >= NWAVE * NCHUNK * 32 * 16) return;
    const int e   = d & 15;
    const int ln  = (d >> 4) & 31;
    const int grp = d >> 9;                            // o_tile*NCHUNK + chunk
    const int kc  = grp % NCHUNK;
    const int ot  = grp / NCHUNK;
    const int n   = ot * 16 + (ln & 15);
    const int k   = kc * 32 + 16 * (ln >> 4) + e;
    Whf[d] = (k < KDIM) ? (_Float16)W[n * KDIM + k] : (_Float16)0.0f;
}

// ---- main kernel -----------------------------------------------------------
__global__ __launch_bounds__(256)
void fgg_conv2d_wmma_f16(const float* __restrict__ x,       // (32,64,64,65) f32
                         const _Float16* __restrict__ Whf,  // fragment-order f16
                         const float* __restrict__ bias,    // (128) f32
                         float* __restrict__ out)           // (32,64,64,129) f32
{
    __shared__ __align__(32) _Float16 hcat[TILE_N * HSTRIDE]; // later: s-tile
    __shared__ float tch[TILE_N];

    const int tid     = threadIdx.x;
    const int lane    = tid & 31;
    const int wv      = tid >> 5;          // wave id = output-channel tile
    const int rowbase = blockIdx.x * TILE_N;

    // ---------------- Phase 1: build f16 hcat tile in LDS ------------------
    // One wave per (row, patch-pos) task; lanes stride channels -> coalesced.
    for (int task = wv; task < TILE_N * 9; task += NWAVE) {
        const int r   = task / 9;
        const int pos = task % 9;
        const int p   = rowbase + r;
        const int ww  = p & 63;
        const int hh  = (p >> 6) & 63;
        const int bb  = p >> 12;
        const int sh  = hh + pos / 3 - 1;
        const int sw  = ww + pos % 3 - 1;
        _Float16* dst = &hcat[r * HSTRIDE + 1 + pos * 64];
        if (sh >= 0 && sh < 64 && sw >= 0 && sw < 64) {
            const float* src = x + ((size_t)((bb * 64 + sh) * 64 + sw) * 65 + 1);
            dst[2 * lane]     = (_Float16)src[2 * lane];
            dst[2 * lane + 1] = (_Float16)src[2 * lane + 1];
        } else {
            dst[2 * lane]     = (_Float16)0.0f;  // zero space padding
            dst[2 * lane + 1] = (_Float16)0.0f;
        }
    }
    // time channel -> t_cat (computed fully in f32), plus K zero-padding
    for (int r = tid; r < TILE_N; r += 256) {
        const int p  = rowbase + r;
        const int ww = p & 63;
        const int hh = (p >> 6) & 63;
        const int bb = p >> 12;
        float ssum = 0.0f;
        #pragma unroll
        for (int pos = 0; pos < 9; ++pos) {
            const int sh = hh + pos / 3 - 1;
            const int sw = ww + pos % 3 - 1;
            float tval = 1.0f;  // sqrt(1/c) boundary padding for time channel
            if (sh >= 0 && sh < 64 && sw >= 0 && sw < 64)
                tval = x[(size_t)((bb * 64 + sh) * 64 + sw) * 65];
            ssum += tval * tval;
        }
        hcat[r * HSTRIDE + 0] = (_Float16)sqrtf(fmaxf(ssum - 8.0f, 1e-7f));
        for (int k = KDIM; k < KPAD; ++k)
            hcat[r * HSTRIDE + k] = (_Float16)0.0f;
    }
    __syncthreads();

    // ---------------- Phase 2: GEMM via v_wmma_f32_16x16x32_f16 ------------
    const int half = lane >> 4;            // K half selector per ISA layout
    const int l    = lane & 15;            // A: M index; B(frag): N index
    // fragment-order B: contiguous coalesced 32B per lane, 1KB per chunk/wave
    const _Float16* Wp = Whf + (((size_t)wv * NCHUNK) * 32 + lane) * 16;

    v8f acc[NTILES] = {};                  // 8 independent accumulator chains

    v16h bfrag = *reinterpret_cast<const v16h*>(Wp);   // prefetch chunk 0

    #pragma unroll 1
    for (int kc = 0; kc < NCHUNK; ++kc) {
        // software pipeline: next chunk's B load overlaps this chunk's WMMAs
        const int kn = (kc + 1 < NCHUNK) ? kc + 1 : kc;
        const v16h bnext = *reinterpret_cast<const v16h*>(Wp + (size_t)kn * 512);
        #pragma unroll
        for (int nt = 0; nt < NTILES; ++nt) {
            // A fragment: VGPR0-7 = K[16h..16h+15] (matches Whf fragment order)
            const _Float16* arow =
                &hcat[(nt * 16 + l) * HSTRIDE + kc * 32 + 16 * half];
            const v8h alo = *reinterpret_cast<const v8h*>(arow);
            const v8h ahi = *reinterpret_cast<const v8h*>(arow + 8);
            const v16h afrag = __builtin_shufflevector(
                alo, ahi, 0, 1, 2, 3, 4, 5, 6, 7, 8, 9, 10, 11, 12, 13, 14, 15);
            acc[nt] = __builtin_amdgcn_wmma_f32_16x16x32_f16(
                false, afrag, false, bfrag, (short)0, acc[nt], false, false);
        }
        bfrag = bnext;
    }

    const float bv = bias[wv * 16 + l];    // each lane holds one out channel

    // ---------------- Phase 3: epilogue ------------------------------------
    __syncthreads();                       // all waves done reading hcat
    float* stile = reinterpret_cast<float*>(hcat);  // reuse LDS: [128][128] f32
    #pragma unroll
    for (int nt = 0; nt < NTILES; ++nt) {
        #pragma unroll
        for (int v = 0; v < 8; ++v) {
            const int m = nt * 16 + v + 8 * half;   // D layout rows
            stile[m * OUTC + wv * 16 + l] = acc[nt][v] + bv;
        }
    }
    __syncthreads();

    for (int r = tid; r < TILE_N; r += 256) {
        float sum = 1.0f;                  // 1/c
        #pragma unroll 8
        for (int j = 0; j < OUTC; ++j) {
            const float sv = stile[r * OUTC + ((j + r) & 127)]; // bank-rotate
            sum += sv * sv;
        }
        tch[r] = sqrtf(fmaxf(sum, 1e-7f));
    }
    __syncthreads();

    // coalesced write of (t, s0..s127) per pixel
    for (int idx = tid; idx < TILE_N * 129; idx += 256) {
        const int r  = idx / 129;
        const int ch = idx % 129;
        const float val = (ch == 0) ? tch[r] : stile[r * OUTC + ch - 1];
        out[(size_t)(rowbase + r) * 129 + ch] = val;
    }
}

extern "C" void kernel_launch(void* const* d_in, const int* in_sizes, int n_in,
                              void* d_out, int out_size, void* d_ws, size_t ws_size,
                              hipStream_t stream) {
    (void)in_sizes; (void)n_in; (void)out_size; (void)ws_size;
    const float* x  = (const float*)d_in[0];
    const float* Wm = (const float*)d_in[1];
    const float* b  = (const float*)d_in[2];
    float* out      = (float*)d_out;
    _Float16* Whf   = (_Float16*)d_ws;     // 8*19*32*16*2 = 155,648 bytes

    const int wtot = NWAVE * NCHUNK * 32 * 16;
    prep_w_frag<<<(wtot + 255) / 256, 256, 0, stream>>>(Wm, Whf);

    const int total_rows = 32 * 64 * 64;   // 131072 pixels
    fgg_conv2d_wmma_f16<<<total_rows / TILE_N, 256, 0, stream>>>(x, Whf, b, out);
}
